// CrossAttention_56203942035821
// MI455X (gfx1250) — matrix operand order repaired and checked by reading, not compiled, not run on hardware
//
#include <hip/hip_runtime.h>
#include <hip/hip_bf16.h>

typedef __attribute__((ext_vector_type(16))) __bf16 v16bf;
typedef __attribute__((ext_vector_type(8)))  float  v8f;

union FragA { v16bf v; uint4 q[2]; };

// -------- async global->LDS path (gfx1250), guarded so compile never breaks --
#if defined(__has_builtin)
#if __has_builtin(__builtin_amdgcn_global_load_async_to_lds_b128) && \
    __has_builtin(__builtin_amdgcn_s_wait_asynccnt)
#define USE_ASYNC_LDS 1
#endif
#endif

#ifdef USE_ASYNC_LDS
typedef __attribute__((__vector_size__(4 * sizeof(int)))) int v4i_vs;
typedef __attribute__((address_space(1))) v4i_vs* gptr128;
typedef __attribute__((address_space(3))) v4i_vs* lptr128;
__device__ __forceinline__ void async_copy16(const void* g, void* l) {
    __builtin_amdgcn_global_load_async_to_lds_b128(
        (gptr128)g, (lptr128)l, 0, 0);
}
__device__ __forceinline__ void async_wait0() {
    __builtin_amdgcn_s_wait_asynccnt(0);
}
#else
__device__ __forceinline__ void async_wait0() {}
#endif

__device__ __forceinline__ unsigned short f2bf(float f) {
    unsigned int u = __float_as_uint(f);
    unsigned int r = u + 0x7FFFu + ((u >> 16) & 1u);
    return (unsigned short)(r >> 16);
}

// ---------------------------------------------------------------------------
// LayerNorm (768 cols) -> bf16
// ---------------------------------------------------------------------------
__global__ __launch_bounds__(256) void ln_bf16_kernel(
    const float* __restrict__ x, const float* __restrict__ w,
    unsigned short* __restrict__ xn)
{
    __shared__ float s_sum[8], s_sq[8];
    const int dim = 768;
    int row = blockIdx.x;
    const float* xr = x + (size_t)row * dim;
    float v[3], sum = 0.f, sq = 0.f;
    #pragma unroll
    for (int k = 0; k < 3; ++k) {
        v[k] = xr[threadIdx.x + k * 256];
        sum += v[k]; sq += v[k] * v[k];
    }
    #pragma unroll
    for (int off = 16; off >= 1; off >>= 1) {
        sum += __shfl_xor(sum, off, 32);
        sq  += __shfl_xor(sq,  off, 32);
    }
    int lane = threadIdx.x & 31, wv = threadIdx.x >> 5;
    if (lane == 0) { s_sum[wv] = sum; s_sq[wv] = sq; }
    __syncthreads();
    sum = 0.f; sq = 0.f;
    #pragma unroll
    for (int i = 0; i < 8; ++i) { sum += s_sum[i]; sq += s_sq[i]; }
    float mu   = sum / dim;
    float var  = sq / dim - mu * mu;
    float rstd = rsqrtf(var + 1e-5f);
    #pragma unroll
    for (int k = 0; k < 3; ++k) {
        int c = threadIdx.x + k * 256;
        xn[(size_t)row * dim + c] = f2bf((v[k] - mu) * rstd * w[c]);
    }
}

// ---------------------------------------------------------------------------
// fp32 -> bf16 elementwise
// ---------------------------------------------------------------------------
__global__ __launch_bounds__(256) void cvt_bf16_kernel(
    const float* __restrict__ src, unsigned short* __restrict__ dst, int n)
{
    int i = blockIdx.x * 256 + threadIdx.x;
    if (i < n) dst[i] = f2bf(src[i]);
}

// ---------------------------------------------------------------------------
// Generic bf16 WMMA GEMM: C[M,N] = alpha * A[M,K] @ B[K,N]
// BM=64 BN=64 BK=32, 128 threads = 4 waves, wave w -> rows [w*16, w*16+16)
// ---------------------------------------------------------------------------
#define GBM 64
#define GBN 64
#define GBK 32

__global__ __launch_bounds__(128) void gemm_bf16_kernel(
    const unsigned short* __restrict__ A, const unsigned short* __restrict__ B,
    void* __restrict__ Cout, int M, int N, int K, float alpha, int out_bf16)
{
    __shared__ unsigned short As[GBM * GBK];   // [64][32] row-major
    __shared__ unsigned short Bt[GBN * GBK];   // [n][k] (B transposed)

    int tid  = threadIdx.x;
    int lane = tid & 31, w = tid >> 5;
    int n0 = blockIdx.x * GBN;
    int m0 = blockIdx.y * GBM;
    int l15 = lane & 15, lhi = lane >> 4;

    v8f acc[4];
    #pragma unroll
    for (int i = 0; i < 4; ++i)
        #pragma unroll
        for (int e = 0; e < 8; ++e) acc[i][e] = 0.f;

    for (int kb = 0; kb < K; kb += GBK) {
        __syncthreads();
        {   // A tile: thread -> (row = t>>1, 16-col half = t&1), 2x 16B
            int r = tid >> 1, half = tid & 1;
            const unsigned short* src = A + (size_t)(m0 + r) * K + kb + half * 16;
            unsigned short* dst = As + r * GBK + half * 16;
#ifdef USE_ASYNC_LDS
            async_copy16(src, dst);           // global -> LDS, no VGPR round trip
            async_copy16(src + 8, dst + 8);
#else
            ((uint4*)dst)[0] = ((const uint4*)src)[0];
            ((uint4*)dst)[1] = ((const uint4*)src)[1];
#endif
        }
        {   // B tile: coalesced row read, transposed LDS write
            int k = tid >> 2, seg = tid & 3;
            const unsigned short* src = B + (size_t)(kb + k) * N + n0 + seg * 16;
            #pragma unroll
            for (int c = 0; c < 16; ++c)
                Bt[(seg * 16 + c) * GBK + k] = src[c];
        }
        async_wait0();
        __syncthreads();

        if (kb + GBK < K)   // gfx1250 global_prefetch_b8 on next A tile
            __builtin_prefetch(A + (size_t)(m0 + (tid >> 1)) * K + kb + GBK, 0, 0);

        FragA a;   // A frag: lane row = l15, K-chunks {kbase..+7, kbase+16..+23}
        int arow = w * 16 + l15;
        int kbase = lhi * 8;
        a.q[0] = *(const uint4*)(As + arow * GBK + kbase);
        a.q[1] = *(const uint4*)(As + arow * GBK + kbase + 16);
        #pragma unroll
        for (int nt = 0; nt < 4; ++nt) {
            FragA b;   // B frag: lane col = l15, K-chunks {boff..+7, boff+16..+23}
            int brow = nt * 16 + l15;
            int boff = lhi * 8;
            b.q[0] = *(const uint4*)(Bt + brow * GBK + boff);
            b.q[1] = *(const uint4*)(Bt + brow * GBK + boff + 16);
            acc[nt] = __builtin_amdgcn_wmma_f32_16x16x32_bf16(
                false, a.v, false, b.v, (short)0, acc[nt], false, false);
        }
    }

    #pragma unroll
    for (int nt = 0; nt < 4; ++nt) {
        int col = n0 + nt * 16 + l15;
        #pragma unroll
        for (int r = 0; r < 8; ++r) {
            int row = m0 + w * 16 + lhi * 8 + r;
            float v = acc[nt][r] * alpha;
            if (out_bf16)
                ((unsigned short*)Cout)[(size_t)row * N + col] = f2bf(v);
            else
                ((float*)Cout)[(size_t)row * N + col] = v;
        }
    }
}

// ---------------------------------------------------------------------------
// Flash attention: one workgroup = (batch, 32-row n-tile), 12 waves = 12 heads
// Multi-query: K/V + mask tiles in LDS shared by all heads (loaded once).
// ---------------------------------------------------------------------------
#define AH 12
#define ANT 32
#define AJT 32

__global__ __launch_bounds__(384) void attn_kernel(
    const unsigned short* __restrict__ Q,   // [b, n, 768] bf16, pre-scaled
    const unsigned short* __restrict__ KV,  // [b, j, 128] bf16 (K|V)
    const float* __restrict__ mask,         // [b, n, j] fp32
    unsigned short* __restrict__ AO,        // [b, n, 768] bf16
    int n, int j)
{
    __shared__ unsigned short Kt[AJT * 64];          // [j][dh]
    __shared__ unsigned short Vt[64 * AJT];          // [dh][j] (transposed)
    __shared__ float          Mt[ANT * AJT];         // mask tile
    __shared__ unsigned short Pl[AH * ANT * AJT];    // per-head P round-trip

    int tid  = threadIdx.x;
    int lane = tid & 31;
    int h    = tid >> 5;                   // wave index == head
    int bb   = blockIdx.y;
    int i0   = blockIdx.x * ANT;
    int l15  = lane & 15, lhi = lane >> 4;

    // Q fragments (registers, reused all j-steps): rows i0..i0+31, cols h*64..+63
    FragA aQ[2][2];
    #pragma unroll
    for (int mt = 0; mt < 2; ++mt)
        #pragma unroll
        for (int kh = 0; kh < 2; ++kh) {
            const unsigned short* src =
                Q + (size_t)(bb * n + i0 + mt * 16 + l15) * 768 + h * 64 + kh * 32 + lhi * 8;
            aQ[mt][kh].q[0] = *(const uint4*)(src);
            aQ[mt][kh].q[1] = *(const uint4*)(src + 16);
        }

    float mst[2][8], lst[2][8];
    v8f o[2][4];
    #pragma unroll
    for (int mt = 0; mt < 2; ++mt)
        #pragma unroll
        for (int r = 0; r < 8; ++r) { mst[mt][r] = -3.0e38f; lst[mt][r] = 0.f; }
    #pragma unroll
    for (int mt = 0; mt < 2; ++mt)
        #pragma unroll
        for (int nd = 0; nd < 4; ++nd)
            #pragma unroll
            for (int e = 0; e < 8; ++e) o[mt][nd][e] = 0.f;

    for (int j0 = 0; j0 < j; j0 += AJT) {
        __syncthreads();
#ifdef USE_ASYNC_LDS
        if (tid < 256) {  // waves 0..7: async K tile (4KB) + mask tile (4KB)
            int jj = tid >> 3, seg = tid & 7;          // 32 rows x 8 chunks
            async_copy16(KV + (size_t)(bb * j + j0 + jj) * 128 + seg * 8,
                         Kt + jj * 64 + seg * 8);
            int r = tid >> 3;                          // mask: 32 rows x 8 chunks
            async_copy16(mask + ((size_t)bb * n + i0 + r) * j + j0 + seg * 4,
                         Mt + r * AJT + seg * 4);
        }
        for (int idx = tid; idx < AJT * 64; idx += 384) {  // V transpose (sync)
            int jj = idx >> 6, dh = idx & 63;
            Vt[dh * AJT + jj] = KV[(size_t)(bb * j + j0 + jj) * 128 + 64 + dh];
        }
        async_wait0();
#else
        for (int idx = tid; idx < AJT * 64; idx += 384) {
            int jj = idx >> 6, dh = idx & 63;
            const unsigned short* kvp = KV + (size_t)(bb * j + j0 + jj) * 128;
            Kt[jj * 64 + dh]  = kvp[dh];
            Vt[dh * AJT + jj] = kvp[64 + dh];
        }
        for (int idx = tid; idx < ANT * AJT; idx += 384) {
            int r = idx >> 5, c = idx & 31;
            Mt[idx] = mask[((size_t)bb * n + i0 + r) * j + j0 + c];
        }
#endif
        __syncthreads();

        // S = Q @ K^T  (M=32, N=32, K=64)
        v8f s[2][2];
        #pragma unroll
        for (int mt = 0; mt < 2; ++mt)
            #pragma unroll
            for (int nt = 0; nt < 2; ++nt) {
                #pragma unroll
                for (int e = 0; e < 8; ++e) s[mt][nt][e] = 0.f;
                #pragma unroll
                for (int kh = 0; kh < 2; ++kh) {
                    FragA bK;
                    int brow = nt * 16 + l15;
                    int boff = kh * 32 + lhi * 8;
                    bK.q[0] = *(const uint4*)(Kt + brow * 64 + boff);
                    bK.q[1] = *(const uint4*)(Kt + brow * 64 + boff + 16);
                    s[mt][nt] = __builtin_amdgcn_wmma_f32_16x16x32_bf16(
                        false, aQ[mt][kh].v, false, bK.v, (short)0, s[mt][nt], false, false);
                }
            }

        // mask + online softmax (row reductions across the 16-lane N group)
        #pragma unroll
        for (int mt = 0; mt < 2; ++mt)
            #pragma unroll
            for (int r = 0; r < 8; ++r) {
                int M = mt * 16 + lhi * 8 + r;
                float s0 = s[mt][0][r] + Mt[M * AJT + l15];
                float s1 = s[mt][1][r] + Mt[M * AJT + 16 + l15];
                float mx = fmaxf(s0, s1);
                #pragma unroll
                for (int off = 1; off <= 8; off <<= 1)
                    mx = fmaxf(mx, __shfl_xor(mx, off, 32));
                float mnew = fmaxf(mst[mt][r], mx);
                float al   = __expf(mst[mt][r] - mnew);
                mst[mt][r] = mnew;
                float p0 = __expf(s0 - mnew);
                float p1 = __expf(s1 - mnew);
                Pl[(h * ANT + M) * AJT + l15]      = f2bf(p0);
                Pl[(h * ANT + M) * AJT + 16 + l15] = f2bf(p1);
                float rs = p0 + p1;
                #pragma unroll
                for (int off = 1; off <= 8; off <<= 1)
                    rs += __shfl_xor(rs, off, 32);
                lst[mt][r] = lst[mt][r] * al + rs;
                #pragma unroll
                for (int nd = 0; nd < 4; ++nd) o[mt][nd][r] *= al;
            }

        // O += P @ V  (M=32, N=64, K=32); P re-read from LDS as A-fragments
        #pragma unroll
        for (int mt = 0; mt < 2; ++mt) {
            FragA aP;
            int prow = mt * 16 + l15;
            aP.q[0] = *(const uint4*)(Pl + (h * ANT + prow) * AJT + lhi * 8);
            aP.q[1] = *(const uint4*)(Pl + (h * ANT + prow) * AJT + lhi * 8 + 16);
            #pragma unroll
            for (int nd = 0; nd < 4; ++nd) {
                FragA bV;
                int vrow = nd * 16 + l15;
                bV.q[0] = *(const uint4*)(Vt + vrow * AJT + lhi * 8);
                bV.q[1] = *(const uint4*)(Vt + vrow * AJT + lhi * 8 + 16);
                o[mt][nd] = __builtin_amdgcn_wmma_f32_16x16x32_bf16(
                    false, aP.v, false, bV.v, (short)0, o[mt][nd], false, false);
            }
        }
    }

    // normalize and emit bf16 [b, n, 768] (head h -> cols h*64..)
    #pragma unroll
    for (int mt = 0; mt < 2; ++mt)
        #pragma unroll
        for (int nd = 0; nd < 4; ++nd) {
            int col = h * 64 + nd * 16 + l15;
            #pragma unroll
            for (int r = 0; r < 8; ++r) {
                int row = i0 + mt * 16 + lhi * 8 + r;
                AO[(size_t)(bb * n + row) * 768 + col] = f2bf(o[mt][nd][r] / lst[mt][r]);
            }
        }
}

// ---------------------------------------------------------------------------
// Host-side orchestration
// ---------------------------------------------------------------------------
extern "C" void kernel_launch(void* const* d_in, const int* in_sizes, int n_in,
                              void* d_out, int out_size, void* d_ws, size_t ws_size,
                              hipStream_t stream) {
    const int B = 4, N = 2048, J = 2048, DIM = 768, INNER = 768, DKV = 128;
    const int ROWS = B * N;               // 8192
    const float SCALE = 0.125f;           // 64^-0.5

    const float* x    = (const float*)d_in[0];
    const float* ctx  = (const float*)d_in[1];
    const float* mask = (const float*)d_in[2];
    const float* ln_w = (const float*)d_in[3];
    const float* Wq   = (const float*)d_in[4];
    const float* Wkv  = (const float*)d_in[5];
    const float* Wo   = (const float*)d_in[6];

    size_t off = 0;
    auto alloc = [&](size_t bytes) {
        size_t o = off; off += (bytes + 255) & ~(size_t)255; return o;
    };
    char* ws = (char*)d_ws;
    unsigned short* xn_bf  = (unsigned short*)(ws + alloc((size_t)ROWS * DIM * 2));
    unsigned short* ctx_bf = (unsigned short*)(ws + alloc((size_t)ROWS * DIM * 2));
    unsigned short* wq_bf  = (unsigned short*)(ws + alloc((size_t)DIM * INNER * 2));
    unsigned short* wkv_bf = (unsigned short*)(ws + alloc((size_t)DIM * DKV * 2));
    unsigned short* wo_bf  = (unsigned short*)(ws + alloc((size_t)INNER * DIM * 2));
    unsigned short* q_bf   = (unsigned short*)(ws + alloc((size_t)ROWS * INNER * 2));
    unsigned short* kv_bf  = (unsigned short*)(ws + alloc((size_t)ROWS * DKV * 2));
    unsigned short* ao_bf  = (unsigned short*)(ws + alloc((size_t)ROWS * INNER * 2));
    (void)ws_size; (void)in_sizes; (void)n_in; (void)out_size;

    // 1) LayerNorm -> bf16; convert context + weights -> bf16
    ln_bf16_kernel<<<ROWS, 256, 0, stream>>>(x, ln_w, xn_bf);
    {
        int nc = ROWS * DIM;
        cvt_bf16_kernel<<<(nc + 255) / 256, 256, 0, stream>>>(ctx, ctx_bf, nc);
        int nq = DIM * INNER;
        cvt_bf16_kernel<<<(nq + 255) / 256, 256, 0, stream>>>(Wq, wq_bf, nq);
        int nkv = DIM * DKV;
        cvt_bf16_kernel<<<(nkv + 255) / 256, 256, 0, stream>>>(Wkv, wkv_bf, nkv);
        int no = INNER * DIM;
        cvt_bf16_kernel<<<(no + 255) / 256, 256, 0, stream>>>(Wo, wo_bf, no);
    }

    // 2) Q = xn @ Wq (scale folded), KV = ctx @ Wkv
    gemm_bf16_kernel<<<dim3(INNER / GBN, ROWS / GBM), 128, 0, stream>>>(
        xn_bf, wq_bf, q_bf, ROWS, INNER, DIM, SCALE, 1);
    gemm_bf16_kernel<<<dim3(DKV / GBN, ROWS / GBM), 128, 0, stream>>>(
        ctx_bf, wkv_bf, kv_bf, ROWS, DKV, DIM, 1.0f, 1);

    // 3) Flash attention (12 waves = 12 heads share K/V + mask tiles)
    attn_kernel<<<dim3(N / ANT, B), 384, 0, stream>>>(
        q_bf, kv_bf, mask, ao_bf, N, J);

    // 4) out = AO @ Wo -> fp32 d_out
    gemm_bf16_kernel<<<dim3(DIM / GBN, ROWS / GBM), 128, 0, stream>>>(
        ao_bf, wo_bf, d_out, ROWS, DIM, INNER, 1.0f, 0);
}
